// Fraser_71313636982862
// MI455X (gfx1250) — matrix operand order repaired and checked by reading, not compiled, not run on hardware
//
#include <hip/hip_runtime.h>
#include <hip/hip_bf16.h>
#include <stdint.h>

typedef __attribute__((ext_vector_type(16))) __bf16 v16bf;
typedef __attribute__((ext_vector_type(8)))  float  v8f;
typedef __attribute__((ext_vector_type(4)))  int    v4i;

#define GSIDE  32
#define BATCH  16384
#define HID    2048
#define NLAYER 8
#define LDSK   72   // padded LDS row stride (elements): 144 B -> conflict-free

__device__ __forceinline__ unsigned short f2bf(float f) {
  unsigned int u = __float_as_uint(f);
  u += 0x7FFFu + ((u >> 16) & 1u);          // round-to-nearest-even
  return (unsigned short)(u >> 16);
}

struct BF16Frag { uint4 lo, hi; };           // 32 bytes == v16bf

// Async global -> LDS copy of 16 bytes (per lane). ASYNCcnt-tracked.
__device__ __forceinline__ void async_cp16(const unsigned short* g, unsigned short* l) {
  typedef __attribute__((address_space(1))) v4i GV4;
  typedef __attribute__((address_space(3))) v4i LV4;
  __builtin_amdgcn_global_load_async_to_lds_b128(
      (GV4*)(uintptr_t)g, (LV4*)(uint32_t)(uintptr_t)l, 0, 0);
}
#define WAIT_ASYNC0() asm volatile("s_wait_asynccnt 0x0" ::: "memory")

// ---------------------------------------------------------------------------
// 1) Patch extraction: find agent cell (==3.0), gather 5x5 patch, pad K to 64,
//    emit bf16 directly (GEMM input precision).
// ---------------------------------------------------------------------------
__global__ __launch_bounds__(256) void patch_kernel(const float* __restrict__ x,
                                                    unsigned short* __restrict__ patch) {
  const int b = blockIdx.x;
  const int t = threadIdx.x;
  __shared__ int sidx;
  if (t == 0) sidx = 1 << 30;
  __syncthreads();
  const float* xb = x + (size_t)b * (GSIDE * GSIDE);
  for (int i = t; i < GSIDE * GSIDE; i += 256)
    if (xb[i] == 3.0f) atomicMin(&sidx, i);
  __syncthreads();
  const int idx = sidx;
  const int ci = idx >> 5, cj = idx & 31;
  if (t < 64) {
    float v = 0.0f;
    if (t < 25) {
      const int pr = ci - 2 + t / 5;
      const int pc = cj - 2 + t % 5;
      v = (pr >= 0 && pr < GSIDE && pc >= 0 && pc < GSIDE) ? xb[pr * GSIDE + pc]
                                                           : -1.0f;
    }
    patch[(size_t)b * 64 + t] = f2bf(v);
  }
}

// ---------------------------------------------------------------------------
// 2a) embed_W [25,2048] f32 -> Wt [2048,64] bf16 (K padded with zeros)
// ---------------------------------------------------------------------------
__global__ __launch_bounds__(256) void wt_embed_kernel(const float* __restrict__ eW,
                                                       unsigned short* __restrict__ Wt) {
  const int idx = blockIdx.x * 256 + threadIdx.x;
  if (idx >= HID * 64) return;
  const int n = idx >> 6, k = idx & 63;
  Wt[idx] = (k < 25) ? f2bf(eW[(size_t)k * HID + n]) : (unsigned short)0;
}

// ---------------------------------------------------------------------------
// 2b) layers_W [8,2048,2048] f32 row-major(K,N) -> Wt [8,2048,2048] bf16 (N,K)
// ---------------------------------------------------------------------------
__global__ __launch_bounds__(256) void wt_layers_kernel(const float* __restrict__ W,
                                                        unsigned short* __restrict__ Wt) {
  __shared__ float t[32][33];
  const int l = blockIdx.z;
  const float* Wl = W + (size_t)l * HID * HID;
  unsigned short* Wtl = Wt + (size_t)l * HID * HID;
  const int k0 = blockIdx.y * 32;
  const int n0 = blockIdx.x * 32;
  const int tx = threadIdx.x, ty = threadIdx.y;   // 32 x 8
#pragma unroll
  for (int j = 0; j < 4; ++j)
    t[ty + j * 8][tx] = Wl[(size_t)(k0 + ty + j * 8) * HID + n0 + tx];
  __syncthreads();
#pragma unroll
  for (int j = 0; j < 4; ++j)
    Wtl[(size_t)(n0 + ty + j * 8) * HID + k0 + tx] = f2bf(t[tx][ty + j * 8]);
}

// ---------------------------------------------------------------------------
// 3) GEMM: fuse(A_bf16[M,Ktot] @ Wt^T, bias, relu, residual_f32)
//    -> out_f32 (residual carry / LN input) and out_bf16 (next layer's A).
//    128x128 block tile, BK=64, double-buffered LDS filled with
//    GLOBAL_LOAD_ASYNC_TO_LDS_B128; 8 waves (2x4), wave tile 64x32 (4x2 WMMA).
//    LDS rows padded to 72 elems (144 B) so 16 consecutive-row b128 reads hit
//    16 distinct bank groups (36*l mod 64 is a permutation for l=0..15).
// ---------------------------------------------------------------------------
__global__ __launch_bounds__(256) void gemm_bf16(
    const unsigned short* __restrict__ A, int Ktot,
    const unsigned short* __restrict__ Wt,
    const float* __restrict__ bias,
    const float* __restrict__ residual,
    float* __restrict__ out_f32,
    unsigned short* __restrict__ out_bf16, int do_relu) {
  __shared__ __align__(16) unsigned short As[2][128][LDSK];  // 2 x 18 KB
  __shared__ __align__(16) unsigned short Bs[2][128][LDSK];  // 2 x 18 KB

  const int tid = threadIdx.x;
  const int m0 = blockIdx.y * 128;
  const int n0 = blockIdx.x * 128;
  const int wid = tid >> 5;
  const int lane = tid & 31;
  const int wm = (wid >> 2) * 64;
  const int wn = (wid & 3) * 32;
  const int l16 = lane & 15;
  const bool hihalf = lane >= 16;
  const int aoff = hihalf ? 8 : 0;    // A: K element offset per ISA layout
  const int boff = hihalf ? 8 : 0;    // B: K element offset per ISA layout

  // this thread's 4 staging chunks per tile (16B each): rows r, col c (elems)
  const int sr = tid >> 3;            // 0..31 base row
  const int sc = (tid & 7) << 3;      // 0..56 col

  v8f acc[4][2] = {};

  const int nkt = Ktot >> 6;

  // prologue: stage K-tile 0 into buffer 0
  {
    const int k0 = 0;
#pragma unroll
    for (int it = 0; it < 4; ++it) {
      const int r = sr + it * 32;
      async_cp16(A  + (size_t)(m0 + r) * Ktot + k0 + sc, &As[0][r][sc]);
      async_cp16(Wt + (size_t)(n0 + r) * Ktot + k0 + sc, &Bs[0][r][sc]);
    }
  }

  int buf = 0;
  for (int kt = 0; kt < nkt; ++kt) {
    WAIT_ASYNC0();
    __syncthreads();                  // buffer `buf` ready on all waves

    if (kt + 1 < nkt) {               // prefetch next K-tile into other buffer
      const int k0 = (kt + 1) << 6;
#pragma unroll
      for (int it = 0; it < 4; ++it) {
        const int r = sr + it * 32;
        async_cp16(A  + (size_t)(m0 + r) * Ktot + k0 + sc, &As[buf ^ 1][r][sc]);
        async_cp16(Wt + (size_t)(n0 + r) * Ktot + k0 + sc, &Bs[buf ^ 1][r][sc]);
      }
    }

#pragma unroll
    for (int ks = 0; ks < 64; ks += 32) {
      v16bf af[4], bfm[2];
#pragma unroll
      for (int mi = 0; mi < 4; ++mi) {
        BF16Frag t;
        t.lo = *(const uint4*)&As[buf][wm + mi * 16 + l16][ks + aoff];
        t.hi = *(const uint4*)&As[buf][wm + mi * 16 + l16][ks + 16 + aoff];
        af[mi] = __builtin_bit_cast(v16bf, t);
      }
#pragma unroll
      for (int ni = 0; ni < 2; ++ni) {
        BF16Frag t;
        t.lo = *(const uint4*)&Bs[buf][wn + ni * 16 + l16][ks + boff];
        t.hi = *(const uint4*)&Bs[buf][wn + ni * 16 + l16][ks + 16 + boff];
        bfm[ni] = __builtin_bit_cast(v16bf, t);
      }
#pragma unroll
      for (int mi = 0; mi < 4; ++mi)
#pragma unroll
        for (int ni = 0; ni < 2; ++ni)
          acc[mi][ni] = __builtin_amdgcn_wmma_f32_16x16x32_bf16(
              false, af[mi], false, bfm[ni], (short)0, acc[mi][ni], false, false);
    }
    buf ^= 1;
  }

  // epilogue: bias (+ relu + residual); C/D layout: lanes0-15 M=vg, 16-31 M=vg+8
#pragma unroll
  for (int mi = 0; mi < 4; ++mi) {
#pragma unroll
    for (int ni = 0; ni < 2; ++ni) {
      const int col = n0 + wn + ni * 16 + l16;
      const int rbase = m0 + wm + mi * 16 + (hihalf ? 8 : 0);
      const float bv = bias[col];
#pragma unroll
      for (int vg = 0; vg < 8; ++vg) {
        const int row = rbase + vg;
        float v = acc[mi][ni][vg] + bv;
        if (do_relu) v = residual[(size_t)row * HID + col] + fmaxf(v, 0.0f);
        out_f32[(size_t)row * HID + col] = v;
        out_bf16[(size_t)row * HID + col] = f2bf(v);
      }
    }
  }
}

// ---------------------------------------------------------------------------
// 4) LayerNorm + head (2048 -> 4), one block per row.
// ---------------------------------------------------------------------------
__global__ __launch_bounds__(256) void ln_head_kernel(
    const float* __restrict__ h, const float* __restrict__ g,
    const float* __restrict__ bb, const float* __restrict__ hw,
    const float* __restrict__ hb, float* __restrict__ out) {
  const int b = blockIdx.x;
  const int t = threadIdx.x;
  const float* row = h + (size_t)b * HID;
  float vals[8];
  float s = 0.f, s2 = 0.f;
#pragma unroll
  for (int j = 0; j < 8; ++j) {
    const float v = row[t + j * 256];
    vals[j] = v;
    s += v;
    s2 += v * v;
  }
  __shared__ float red[256];
  red[t] = s;
  __syncthreads();
  for (int o = 128; o > 0; o >>= 1) {
    if (t < o) red[t] += red[t + o];
    __syncthreads();
  }
  const float mu = red[0] * (1.0f / HID);
  __syncthreads();
  red[t] = s2;
  __syncthreads();
  for (int o = 128; o > 0; o >>= 1) {
    if (t < o) red[t] += red[t + o];
    __syncthreads();
  }
  const float var = red[0] * (1.0f / HID) - mu * mu;
  const float rs = rsqrtf(var + 1e-5f);
  __syncthreads();

  float acc[4] = {0.f, 0.f, 0.f, 0.f};
#pragma unroll
  for (int j = 0; j < 8; ++j) {
    const int e = t + j * 256;
    const float hn = (vals[j] - mu) * rs * g[e] + bb[e];
#pragma unroll
    for (int a = 0; a < 4; ++a) acc[a] += hn * hw[e * 4 + a];
  }
  for (int a = 0; a < 4; ++a) {
    red[t] = acc[a];
    __syncthreads();
    for (int o = 128; o > 0; o >>= 1) {
      if (t < o) red[t] += red[t + o];
      __syncthreads();
    }
    if (t == 0) out[(size_t)b * 4 + a] = red[0] + hb[a];
    __syncthreads();
  }
}

// ---------------------------------------------------------------------------
extern "C" void kernel_launch(void* const* d_in, const int* in_sizes, int n_in,
                              void* d_out, int out_size, void* d_ws, size_t ws_size,
                              hipStream_t stream) {
  (void)in_sizes; (void)n_in; (void)out_size; (void)ws_size;
  const float* x        = (const float*)d_in[0];
  const float* embed_W  = (const float*)d_in[1];
  const float* embed_b  = (const float*)d_in[2];
  const float* layers_W = (const float*)d_in[3];
  const float* layers_b = (const float*)d_in[4];
  const float* ln_g     = (const float*)d_in[5];
  const float* ln_b     = (const float*)d_in[6];
  const float* head_W   = (const float*)d_in[7];
  const float* head_b   = (const float*)d_in[8];
  float* out = (float*)d_out;

  char* ws = (char*)d_ws;
  unsigned short* patchb = (unsigned short*)ws;                          //   2 MiB
  unsigned short* wte    = (unsigned short*)(ws + 2097152ull);           // 256 KiB
  unsigned short* wtl    = (unsigned short*)(ws + 2359296ull);           //  64 MiB
  float*          hAf    = (float*)(ws + 69468160ull);                   // 128 MiB
  float*          hBf    = (float*)(ws + 203685888ull);                  // 128 MiB
  unsigned short* hAb    = (unsigned short*)(ws + 337903616ull);         //  64 MiB
  unsigned short* hBb    = (unsigned short*)(ws + 405012480ull);         //  64 MiB

  patch_kernel<<<BATCH, 256, 0, stream>>>(x, patchb);
  wt_embed_kernel<<<(HID * 64) / 256, 256, 0, stream>>>(embed_W, wte);
  wt_layers_kernel<<<dim3(HID / 32, HID / 32, NLAYER), dim3(32, 8), 0, stream>>>(
      layers_W, wtl);

  const dim3 ggrid(HID / 128, BATCH / 128);
  gemm_bf16<<<ggrid, 256, 0, stream>>>(patchb, 64, wte, embed_b, nullptr,
                                       hAf, hAb, 0);

  float* hinf = hAf;  unsigned short* hinb = hAb;
  float* houtf = hBf; unsigned short* houtb = hBb;
  for (int l = 0; l < NLAYER; ++l) {
    gemm_bf16<<<ggrid, 256, 0, stream>>>(hinb, HID, wtl + (size_t)l * HID * HID,
                                         layers_b + (size_t)l * HID, hinf,
                                         houtf, houtb, 1);
    float* tf = hinf; hinf = houtf; houtf = tf;
    unsigned short* tb = hinb; hinb = houtb; houtb = tb;
  }
  ln_head_kernel<<<BATCH, 256, 0, stream>>>(hinf, ln_g, ln_b, head_W, head_b, out);
}
